// WLS_net_FG_v2_56556129353925
// MI455X (gfx1250) — compile-verified, matrix operand order checked by reading, not run on hardware
//
#include <hip/hip_runtime.h>
#include <hip/hip_bf16.h>

// ---------------------------------------------------------------------------
// KPN denoiser for MI455X (gfx1250) — bf16 WMMA implicit-GEMM convs with
// LDS-staged activation windows.
//   x:(4,10,256,256) -> conv10->100 -> 4x residual(100->100,relu,100->100)
//   -> conv100->441 -> sigmoid - mean -> 21x21 dynamic filter on x[:, :3]
//
// Conv blocking: 256-thread block = 64 pixels x 128 outch.
//   * 3x66xCpad halo window staged once in LDS via async global->LDS loads
//     (zero-filled borders -> branch-free inner loop, EXEC all-ones).
//   * wave = 32 px x 32 outch (2x2 WMMA tiles): per k-step 2 B frags (LDS) +
//     2 A frags (global) feed 4 WMMAs -> 1 global + 1 ds load per WMMA.
// ---------------------------------------------------------------------------

typedef __attribute__((ext_vector_type(16))) __bf16 v16bf;
typedef __attribute__((ext_vector_type(8)))  __bf16 v8bf;
typedef __attribute__((ext_vector_type(8)))  float  v8f;
typedef int v4i_gl __attribute__((vector_size(16)));   // async builtin pointee

#define NB   4
#define HH   256
#define WW   256
#define HW   (HH*WW)
#define NP   (NB*HW)          // 262144 pixels
#define CIN  10
#define CINP 32               // padded input channels
#define CI   100
#define CIP  128              // padded hidden channels
#define KSQ  441
#define KSQP 512              // padded filter channels (multiple of 128)
#define PK   21
#define RAD  10

#if defined(__HIP_DEVICE_COMPILE__) && __has_builtin(__builtin_amdgcn_global_load_async_to_lds_b128)
#define HAVE_ASYNC_LDS 1
#else
#define HAVE_ASYNC_LDS 0
#endif

union Frag { v16bf v; v8bf h[2]; };

__device__ __forceinline__ v8bf zero8() {
    v8bf z;
#pragma unroll
    for (int i = 0; i < 8; ++i) z[i] = (__bf16)0.0f;
    return z;
}

// ---------------------------------------------------------------------------
// LDS-staged implicit-GEMM 3x3 SAME conv.
//   in  : [NP][CinP]  bf16 (NHWC, channel-padded)
//   wgt : [9][Coutp][CinP] bf16
//   bias: [Coutp] f32,  res: optional residual, same layout as out
//   out : [NP][Coutp] bf16
// ACT: 0 none, 1 relu, 2 sigmoid.
// Grid: x = NP/64, y = Coutp/128.  Block 256 = 8 waves (2 nsub x 4 msub).
// ---------------------------------------------------------------------------
template <int CINPAD, int ACT, bool RES>
__global__ __launch_bounds__(256) void conv3x3_wmma(
    const __bf16* __restrict__ in, const __bf16* __restrict__ wgt,
    const float* __restrict__ bias, const __bf16* __restrict__ res,
    __bf16* __restrict__ out, int Coutp)
{
    constexpr int PSTR = CINPAD + 8;            // LDS pixel stride (bank-spread)
    __shared__ __bf16 sact[3 * 66 * PSTR];      // 3 rows x 66 cols halo window

    const int tid  = threadIdx.x;
    const int lane = tid & 31;
    const int wave = tid >> 5;

    const int p0  = blockIdx.x * 64;            // first pixel of this block
    const int b   = p0 / HW;
    const int hw0 = p0 % HW;
    const int y0  = hw0 / WW;
    const int x0  = hw0 % WW;                   // multiple of 64, row-aligned

    // ---- stage 3x66xCpad halo window into LDS (zero-filled at borders) ----
    {
        const int nchunk = 3 * 66 * (CINPAD / 8);
        for (int idx = tid; idx < nchunk; idx += 256) {
            const int k8 = idx % (CINPAD / 8);
            const int t  = idx / (CINPAD / 8);
            const int c  = t % 66;
            const int r  = t / 66;
            const int ys = y0 + r - 1;
            const int xs = x0 + c - 1;
            __bf16* dst = &sact[(r * 66 + c) * PSTR + k8 * 8];
            const bool vld = ((unsigned)ys < (unsigned)HH) && ((unsigned)xs < (unsigned)WW);
#if HAVE_ASYNC_LDS
            if (vld) {
                const __bf16* src = in + ((size_t)b * HW + (size_t)ys * WW + xs) * CINPAD + k8 * 8;
                __builtin_amdgcn_global_load_async_to_lds_b128(
                    (v4i_gl*)src, (v4i_gl*)dst, 0, 0);
            } else {
                *(v8bf*)dst = zero8();
            }
#else
            v8bf v = zero8();
            if (vld)
                v = *(const v8bf*)(in + ((size_t)b * HW + (size_t)ys * WW + xs) * CINPAD + k8 * 8);
            *(v8bf*)dst = v;
#endif
        }
    }
#if HAVE_ASYNC_LDS
#if __has_builtin(__builtin_amdgcn_s_wait_asynccnt)
    __builtin_amdgcn_s_wait_asynccnt(0);
#else
    asm volatile("s_wait_asynccnt 0x0" ::: "memory");
#endif
#endif
    __syncthreads();

    // ---- per-wave tile: 32 pixels x 32 out-channels (2x2 WMMA tiles) ----
    const int n    = lane & 15;                 // A/B row within a 16-tile
    const int hi   = lane >> 4;                 // K sub-block selector
    const int nsub = wave & 1;                  // pixel half (0..1): 32 px each
    const int msub = wave >> 1;                 // out-ch quarter (0..3): 32 co
    const int co0  = blockIdx.y * 128 + msub * 32;
    const int px0  = nsub * 32;                 // wave's first pixel in block

    v8f acc[2][2];                              // [pixel tile][co tile]
#pragma unroll
    for (int mt = 0; mt < 2; ++mt) {
        const float* bptr = bias + co0 + mt * 16 + hi * 8;
#pragma unroll
        for (int r = 0; r < 8; ++r) { acc[0][mt][r] = bptr[r]; acc[1][mt][r] = bptr[r]; }
    }

    const __bf16* wbase = wgt + (size_t)(co0 + n) * CINPAD;   // A row for this lane

    for (int tap = 0; tap < 9; ++tap) {
        const int rr   = tap / 3;               // window row (dy+1)
        const int cofs = tap % 3;               // window col offset (dx+1)
        const __bf16* srow0 = &sact[(rr * 66 + px0 + n + cofs) * PSTR];
        const __bf16* srow1 = srow0 + 16 * PSTR;
        const __bf16* arow  = wbase + (size_t)tap * Coutp * CINPAD;
#pragma unroll
        for (int kk = 0; kk < CINPAD / 32; ++kk) {
            const int kb = kk * 32 + hi * 8;
            Frag B0, B1, A0, A1;
            B0.h[0] = *(const v8bf*)(srow0 + kb);          // ds_load_b128
            B0.h[1] = *(const v8bf*)(srow0 + kb + 16);
            B1.h[0] = *(const v8bf*)(srow1 + kb);
            B1.h[1] = *(const v8bf*)(srow1 + kb + 16);
            A0.h[0] = *(const v8bf*)(arow + kb);           // global_load_b128
            A0.h[1] = *(const v8bf*)(arow + kb + 16);
            A1.h[0] = *(const v8bf*)(arow + 16 * CINPAD + kb);
            A1.h[1] = *(const v8bf*)(arow + 16 * CINPAD + kb + 16);
            acc[0][0] = __builtin_amdgcn_wmma_f32_16x16x32_bf16(
                false, A0.v, false, B0.v, (short)0, acc[0][0], false, false);
            acc[0][1] = __builtin_amdgcn_wmma_f32_16x16x32_bf16(
                false, A1.v, false, B0.v, (short)0, acc[0][1], false, false);
            acc[1][0] = __builtin_amdgcn_wmma_f32_16x16x32_bf16(
                false, A0.v, false, B1.v, (short)0, acc[1][0], false, false);
            acc[1][1] = __builtin_amdgcn_wmma_f32_16x16x32_bf16(
                false, A1.v, false, B1.v, (short)0, acc[1][1], false, false);
        }
    }

    // ---- epilogue: D element (m = r + 8*hi, n) -> 16B bf16 store per tile --
#pragma unroll
    for (int nt = 0; nt < 2; ++nt) {
        const int p = p0 + px0 + nt * 16 + n;   // global pixel for this lane
#pragma unroll
        for (int mt = 0; mt < 2; ++mt) {
            const size_t obase = (size_t)p * Coutp + co0 + mt * 16 + hi * 8;
            v8bf rv;
            if (RES) rv = *(const v8bf*)(res + obase);
            v8bf ov;
#pragma unroll
            for (int r = 0; r < 8; ++r) {
                float v = acc[nt][mt][r];
                if (ACT == 1) v = v > 0.0f ? v : 0.0f;
                if (ACT == 2) v = 1.0f / (1.0f + __expf(-v));
                if (RES) v += (float)rv[r];
                ov[r] = (__bf16)v;
            }
            *(v8bf*)(out + obase) = ov;
        }
    }
}

// ---------------------------------------------------------------------------
// Packing kernels: f32 NCHW / OIHW -> bf16 channel-padded layouts.
// ---------------------------------------------------------------------------
__global__ __launch_bounds__(256) void pack_input(
    const float* __restrict__ x, __bf16* __restrict__ xin)
{
    const int p = blockIdx.x * 256 + threadIdx.x;
    if (p >= NP) return;
    const int b = p / HW, hw = p % HW;
    __bf16* d = xin + (size_t)p * CINP;
#pragma unroll
    for (int c = 0; c < CINP; ++c)
        d[c] = (__bf16)(c < CIN ? x[((size_t)b * CIN + c) * HW + hw] : 0.0f);
}

// src f32 [Cout][Cin][3][3] -> dst bf16 [9][Coutp][Cinp], zero padded.
__global__ __launch_bounds__(256) void pack_weights(
    const float* __restrict__ src, __bf16* __restrict__ dst,
    int Cout, int Cin, int Coutp, int Cinp)
{
    const int idx = blockIdx.x * 256 + threadIdx.x;
    const int total = 9 * Coutp * Cinp;
    if (idx >= total) return;
    const int ci  = idx % Cinp;
    const int t   = idx / Cinp;
    const int co  = t % Coutp;
    const int tap = t / Coutp;
    float v = 0.0f;
    if (co < Cout && ci < Cin) v = src[((size_t)co * Cin + ci) * 9 + tap];
    dst[idx] = (__bf16)v;
}

__global__ __launch_bounds__(256) void pack_bias(
    const float* __restrict__ src, float* __restrict__ dst, int C, int Cp)
{
    const int i = blockIdx.x * 256 + threadIdx.x;
    if (i >= Cp) return;
    dst[i] = (i < C) ? src[i] : 0.0f;
}

// ---------------------------------------------------------------------------
// KPN tail: Wm[p][k] (sigmoid already applied, bf16) -> subtract per-pixel
// mean over 441 taps, apply 21x21 dynamic filter to x[:, :3], add color skip.
// ---------------------------------------------------------------------------
__global__ __launch_bounds__(256) void kpn_apply(
    const float* __restrict__ x, const __bf16* __restrict__ Wm,
    float* __restrict__ out)
{
    const int p = blockIdx.x * 256 + threadIdx.x;
    if (p >= NP) return;
    const int b = p / HW, hw = p % HW;
    const int y = hw / WW, xw = hw % WW;
    const __bf16* wrow = Wm + (size_t)p * KSQP;

    float s = 0.0f;
    for (int k = 0; k < KSQ; ++k) s += (float)wrow[k];
    const float mean = s * (1.0f / (float)KSQ);

    const float* xb = x + (size_t)b * CIN * HW;
    float a0 = 0.0f, a1 = 0.0f, a2 = 0.0f;
    for (int k = 0; k < KSQ; ++k) {
        const int ys = y + k / PK - RAD;
        const int xs = xw + k % PK - RAD;
        if ((unsigned)ys < (unsigned)HH && (unsigned)xs < (unsigned)WW) {
            const float wk = (float)wrow[k] - mean;
            const int off = ys * WW + xs;
            a0 += wk * xb[off];
            a1 += wk * xb[HW + off];
            a2 += wk * xb[2 * HW + off];
        }
    }
    float* ob = out + (size_t)b * 3 * HW + hw;
    ob[0]      = a0 + xb[hw];
    ob[HW]     = a1 + xb[HW + hw];
    ob[2 * HW] = a2 + xb[2 * HW + hw];
}

// ---------------------------------------------------------------------------
extern "C" void kernel_launch(void* const* d_in, const int* in_sizes, int n_in,
                              void* d_out, int out_size, void* d_ws, size_t ws_size,
                              hipStream_t stream)
{
    (void)in_sizes; (void)n_in; (void)out_size; (void)ws_size;
    const float* x   = (const float*)d_in[0];
    // d_in[1] = ref (unused by forward)
    const float* w0  = (const float*)d_in[2];
    const float* b0  = (const float*)d_in[3];
    const float* rw1 = (const float*)d_in[4];
    const float* rb1 = (const float*)d_in[5];
    const float* rw2 = (const float*)d_in[6];
    const float* rb2 = (const float*)d_in[7];
    const float* wf  = (const float*)d_in[8];
    const float* bfb = (const float*)d_in[9];
    float* out = (float*)d_out;

    // Workspace carve-up (256B aligned slices).
    char* ws = (char*)d_ws;
    size_t off = 0;
    auto carve = [&](size_t bytes) { char* p = ws + off; off += (bytes + 255) & ~(size_t)255; return p; };
    __bf16* xin  = (__bf16*)carve((size_t)NP * CINP * 2);        //  16.8 MB
    __bf16* actA = (__bf16*)carve((size_t)NP * CIP * 2);         //  67.1 MB
    __bf16* actB = (__bf16*)carve((size_t)NP * CIP * 2);
    __bf16* actC = (__bf16*)carve((size_t)NP * CIP * 2);
    __bf16* Wm   = (__bf16*)carve((size_t)NP * KSQP * 2);        // 268.4 MB
    __bf16* w0p  = (__bf16*)carve((size_t)9 * CIP * CINP * 2);
    __bf16* rw1p = (__bf16*)carve((size_t)NB * 9 * CIP * CIP * 2);
    __bf16* rw2p = (__bf16*)carve((size_t)NB * 9 * CIP * CIP * 2);
    __bf16* wfp  = (__bf16*)carve((size_t)9 * KSQP * CIP * 2);
    float*  b0p  = (float*)carve(CIP * 4);
    float*  rb1p = (float*)carve((size_t)NB * CIP * 4);
    float*  rb2p = (float*)carve((size_t)NB * CIP * 4);
    float*  bfp  = (float*)carve(KSQP * 4);

    // --- pack inputs / weights / biases into bf16 padded layouts -----------
    pack_input<<<NP / 256, 256, 0, stream>>>(x, xin);
    pack_weights<<<(9 * CIP * CINP + 255) / 256, 256, 0, stream>>>(w0, w0p, CI, CIN, CIP, CINP);
    for (int i = 0; i < NB; ++i) {
        pack_weights<<<(9 * CIP * CIP + 255) / 256, 256, 0, stream>>>(
            rw1 + (size_t)i * CI * CI * 9, rw1p + (size_t)i * 9 * CIP * CIP, CI, CI, CIP, CIP);
        pack_weights<<<(9 * CIP * CIP + 255) / 256, 256, 0, stream>>>(
            rw2 + (size_t)i * CI * CI * 9, rw2p + (size_t)i * 9 * CIP * CIP, CI, CI, CIP, CIP);
        pack_bias<<<1, 256, 0, stream>>>(rb1 + (size_t)i * CI, rb1p + (size_t)i * CIP, CI, CIP);
        pack_bias<<<1, 256, 0, stream>>>(rb2 + (size_t)i * CI, rb2p + (size_t)i * CIP, CI, CIP);
    }
    pack_weights<<<(9 * KSQP * CIP + 255) / 256, 256, 0, stream>>>(wf, wfp, KSQ, CI, KSQP, CIP);
    pack_bias<<<1, 256, 0, stream>>>(b0, b0p, CI, CIP);
    pack_bias<<<2, 256, 0, stream>>>(bfb, bfp, KSQ, KSQP);

    // --- network: LDS-staged WMMA implicit-GEMM convs ----------------------
    const dim3 blk(256);
    const dim3 gHid(NP / 64, CIP / 128);     // 4096 x 1
    const dim3 gFin(NP / 64, KSQP / 128);    // 4096 x 4

    // f = conv0(x) + b0
    conv3x3_wmma<CINP, 0, false><<<gHid, blk, 0, stream>>>(xin, w0p, b0p, nullptr, actA, CIP);

    // 4 residual blocks, ping-pong actA <-> actC with actB as scratch
    __bf16* h = actA;
    __bf16* other = actC;
    for (int i = 0; i < NB; ++i) {
        conv3x3_wmma<CIP, 1, false><<<gHid, blk, 0, stream>>>(
            h, rw1p + (size_t)i * 9 * CIP * CIP, rb1p + (size_t)i * CIP, nullptr, actB, CIP);
        conv3x3_wmma<CIP, 0, true><<<gHid, blk, 0, stream>>>(
            actB, rw2p + (size_t)i * 9 * CIP * CIP, rb2p + (size_t)i * CIP, h, other, CIP);
        __bf16* t = h; h = other; other = t;
    }

    // Wm = sigmoid(conv_f(h))
    conv3x3_wmma<CIP, 2, false><<<gFin, blk, 0, stream>>>(h, wfp, bfp, nullptr, Wm, KSQP);

    // dynamic 21x21 filter + skip connection
    kpn_apply<<<NP / 256, 256, 0, stream>>>(x, Wm, out);
}
